// VirtualScalar_12841952215344
// MI455X (gfx1250) — matrix operand order repaired and checked by reading, not compiled, not run on hardware
//
#include <hip/hip_runtime.h>
#include <hip/hip_bf16.h>

typedef __attribute__((ext_vector_type(16))) __bf16 v16bf;
typedef __attribute__((ext_vector_type(8)))  __bf16 v8bf;
typedef __attribute__((ext_vector_type(8)))  float  v8f;

#define SSCALE (1.0f / 0.6f)

__device__ __forceinline__ __bf16 f2bf(float f) {
    union { float f; unsigned u; } v; v.f = f;
    unsigned r = (v.u + 0x7FFFu + ((v.u >> 16) & 1u)) >> 16;
    unsigned short s = (unsigned short)r;
    return __builtin_bit_cast(__bf16, s);
}

__device__ __forceinline__ float ssilu(float v) {
    return v * (1.0f / (1.0f + __expf(-v))) * SSCALE;
}

// ---------------------------------------------------------------------------
// Pack weights [K x 256] f32 row-major -> bf16 WMMA B-fragment tiles.
// Tile t = kc*16 + ct holds, for lane l (n = l&15, half = l>>4), 16 contiguous
// bf16: W[kc*32 + half*8 + (j&7) + (j>>3)*16][ct*16 + n], j = 0..15  (B layout:
// lanes 0-15 = K 0-7,16-23; lanes 16-31 = K 8-15,24-31 of the chunk).
// ---------------------------------------------------------------------------
__global__ __launch_bounds__(256) void pack_w(const float* __restrict__ W,
                                              __bf16* __restrict__ dst, int KC) {
    int e = blockIdx.x * 256 + threadIdx.x;
    int total = KC * 16 * 512;
    if (e >= total) return;
    int j    = e & 15;
    int lane = (e >> 4) & 31;
    int t    = e >> 9;
    int ct   = t & 15;
    int kc   = t >> 4;
    int hlf  = lane >> 4;
    int n    = lane & 15;
    int k    = kc * 32 + hlf * 8 + (j & 7) + ((j >> 3) << 4);
    dst[e] = f2bf(W[(size_t)k * 256 + ct * 16 + n]);
}

__global__ __launch_bounds__(256) void zero_f32(float* __restrict__ p, int n) {
    int i = blockIdx.x * 256 + threadIdx.x;
    if (i < n) p[i] = 0.0f;
}

// ---------------------------------------------------------------------------
// Fused node MLP: h = ssilu(ssilu([x, vx[batch]]@W1 + b1)@W2 + b2) + x
// plus segment-sum accumulation for the pooling stage.
// Block: 256 threads = 8 waves; 32 rows x 256 cols per block.
// Wave w: row strip rs = w>>2 (16 rows), col group cg = w&3 (64 cols, 4 tiles).
// ---------------------------------------------------------------------------
__global__ __launch_bounds__(256) void node_mlp(
    const float* __restrict__ x, const int* __restrict__ batch,
    const float* __restrict__ vx,
    const __bf16* __restrict__ W1p, const float* __restrict__ b1,
    const __bf16* __restrict__ W2p, const float* __restrict__ b2,
    float* __restrict__ hout, float* __restrict__ sums, int N) {

    __shared__ alignas(16) __bf16 Ain[32 * 520];  // 32 rows x 512 (+8 pad) bf16
    __shared__ alignas(16) __bf16 H1 [32 * 264];  // 32 rows x 256 (+8 pad) bf16
    __shared__ int segs[32];
    float* AliasF32 = (float*)Ain;                // reused after layer-1: 32 x 260 f32

    const int tid = threadIdx.x;
    const int r0  = blockIdx.x * 32;

    if (tid < 32) {
        int gr = r0 + tid; if (gr >= N) gr = N - 1;
        segs[tid] = batch[gr];
    }
    __syncthreads();

    // Stage concat(x, vx[batch]) as bf16 (coalesced).
    for (int idx = tid; idx < 32 * 512; idx += 256) {
        int row = idx >> 9, k = idx & 511;
        int gr = r0 + row; if (gr >= N) gr = N - 1;
        float v = (k < 256) ? x[(size_t)gr * 256 + k]
                            : vx[(size_t)segs[row] * 256 + (k - 256)];
        Ain[row * 520 + k] = f2bf(v);
    }
    __syncthreads();

    const int lane = tid & 31;
    const int w    = tid >> 5;
    const int rs   = w >> 2;      // 0..1
    const int cg   = w & 3;       // 0..3
    const int hlf  = lane >> 4;
    const int m    = lane & 15;

    v8f acc[4];
    for (int t = 0; t < 4; ++t)
        for (int r = 0; r < 8; ++r) acc[t][r] = 0.0f;

    // ---- Layer 1: K = 512 (16 chunks of 32) ----
    for (int kc = 0; kc < 16; ++kc) {
        const __bf16* ap = &Ain[(rs * 16 + m) * 520 + kc * 32 + hlf * 8];
        v8bf a0 = *(const v8bf*)ap;
        v8bf a1 = *(const v8bf*)(ap + 16);
        v16bf a;
        for (int j = 0; j < 8; ++j) { a[j] = a0[j]; a[j + 8] = a1[j]; }
        for (int ct = 0; ct < 4; ++ct) {
            int t = kc * 16 + (cg * 4 + ct);
            v16bf b = *(const v16bf*)(W1p + (size_t)t * 512 + lane * 16);
            acc[ct] = __builtin_amdgcn_wmma_f32_16x16x32_bf16(
                false, a, false, b, (short)0, acc[ct], false, false);
        }
    }
    for (int ct = 0; ct < 4; ++ct) {
        int col = cg * 64 + ct * 16 + m;
        float bias = b1[col];
        for (int r = 0; r < 8; ++r) {
            int row = rs * 16 + hlf * 8 + r;
            H1[row * 264 + col] = f2bf(ssilu(acc[ct][r] + bias));
        }
    }
    __syncthreads();

    // ---- Layer 2: K = 256 (8 chunks) ----
    for (int t = 0; t < 4; ++t)
        for (int r = 0; r < 8; ++r) acc[t][r] = 0.0f;
    for (int kc = 0; kc < 8; ++kc) {
        const __bf16* ap = &H1[(rs * 16 + m) * 264 + kc * 32 + hlf * 8];
        v8bf a0 = *(const v8bf*)ap;
        v8bf a1 = *(const v8bf*)(ap + 16);
        v16bf a;
        for (int j = 0; j < 8; ++j) { a[j] = a0[j]; a[j + 8] = a1[j]; }
        for (int ct = 0; ct < 4; ++ct) {
            int t = kc * 16 + (cg * 4 + ct);
            v16bf b = *(const v16bf*)(W2p + (size_t)t * 512 + lane * 16);
            acc[ct] = __builtin_amdgcn_wmma_f32_16x16x32_bf16(
                false, a, false, b, (short)0, acc[ct], false, false);
        }
    }
    // bias + ssilu + residual -> global h, and f32 copy into LDS for pooling.
    for (int ct = 0; ct < 4; ++ct) {
        int col = cg * 64 + ct * 16 + m;
        float bias = b2[col];
        for (int r = 0; r < 8; ++r) {
            int row = rs * 16 + hlf * 8 + r;
            int gr = r0 + row;
            if (gr < N) {
                size_t g = (size_t)gr * 256 + col;
                float hv = ssilu(acc[ct][r] + bias) + x[g];
                hout[g] = hv;
                AliasF32[row * 260 + col] = hv;
            } else {
                AliasF32[row * 260 + col] = 0.0f;
            }
        }
    }
    __syncthreads();

    // ---- Segment-sum reduction (batch sorted -> ~1 segment per tile) ----
    {
        int col = tid;  // 256 threads, one column each
        float run = 0.0f;
        int cur = segs[0];
        for (int row = 0; row < 32; ++row) {
            if (r0 + row >= N) break;
            int s = segs[row];
            if (s != cur) {
                atomicAdd(&sums[cur * 256 + col], run);
                run = 0.0f; cur = s;
            }
            run += AliasF32[row * 260 + col];
        }
        atomicAdd(&sums[cur * 256 + col], run);
    }
}

// ---------------------------------------------------------------------------
// Counts + pooled mean + bf16 staging of concat(pooled, vx). Single block.
// ---------------------------------------------------------------------------
__global__ __launch_bounds__(256) void pool_stage(
    const int* __restrict__ batch, const float* __restrict__ sums,
    const float* __restrict__ vx, __bf16* __restrict__ Avx, int N, int B) {
    __shared__ int hist[64];
    int tid = threadIdx.x;
    if (tid < B) hist[tid] = 0;
    __syncthreads();
    for (int i = tid; i < N; i += 256) atomicAdd(&hist[batch[i]], 1);
    __syncthreads();
    for (int idx = tid; idx < B * 512; idx += 256) {
        int row = idx >> 9, k = idx & 511;
        float v;
        if (k < 256) {
            float c = (float)hist[row]; if (c < 1.0f) c = 1.0f;
            v = sums[row * 256 + k] / c;
        } else {
            v = vx[row * 256 + (k - 256)];
        }
        Avx[idx] = f2bf(v);
    }
}

// ---------------------------------------------------------------------------
// Virtual-node MLP: vx_new = ssilu(ssilu([pooled, vx]@Wv1 + bv1)@Wv2 + bv2)
// One block of 512 threads = 16 waves; 64 rows x 256 cols exactly.
// ---------------------------------------------------------------------------
__global__ __launch_bounds__(512) void vn_mlp(
    const __bf16* __restrict__ Avx,
    const __bf16* __restrict__ W1p, const float* __restrict__ b1,
    const __bf16* __restrict__ W2p, const float* __restrict__ b2,
    float* __restrict__ out) {

    __shared__ alignas(16) __bf16 H1[64 * 264];

    const int tid  = threadIdx.x;
    const int lane = tid & 31;
    const int w    = tid >> 5;    // 0..15
    const int rs   = w >> 2;      // 0..3
    const int cg   = w & 3;       // 0..3
    const int hlf  = lane >> 4;
    const int m    = lane & 15;

    v8f acc[4];
    for (int t = 0; t < 4; ++t)
        for (int r = 0; r < 8; ++r) acc[t][r] = 0.0f;

    for (int kc = 0; kc < 16; ++kc) {
        const __bf16* ap = Avx + (size_t)(rs * 16 + m) * 512 + kc * 32 + hlf * 8;
        v8bf a0 = *(const v8bf*)ap;
        v8bf a1 = *(const v8bf*)(ap + 16);
        v16bf a;
        for (int j = 0; j < 8; ++j) { a[j] = a0[j]; a[j + 8] = a1[j]; }
        for (int ct = 0; ct < 4; ++ct) {
            int t = kc * 16 + (cg * 4 + ct);
            v16bf b = *(const v16bf*)(W1p + (size_t)t * 512 + lane * 16);
            acc[ct] = __builtin_amdgcn_wmma_f32_16x16x32_bf16(
                false, a, false, b, (short)0, acc[ct], false, false);
        }
    }
    for (int ct = 0; ct < 4; ++ct) {
        int col = cg * 64 + ct * 16 + m;
        float bias = b1[col];
        for (int r = 0; r < 8; ++r) {
            int row = rs * 16 + hlf * 8 + r;
            H1[row * 264 + col] = f2bf(ssilu(acc[ct][r] + bias));
        }
    }
    __syncthreads();

    for (int t = 0; t < 4; ++t)
        for (int r = 0; r < 8; ++r) acc[t][r] = 0.0f;
    for (int kc = 0; kc < 8; ++kc) {
        const __bf16* ap = &H1[(rs * 16 + m) * 264 + kc * 32 + hlf * 8];
        v8bf a0 = *(const v8bf*)ap;
        v8bf a1 = *(const v8bf*)(ap + 16);
        v16bf a;
        for (int j = 0; j < 8; ++j) { a[j] = a0[j]; a[j + 8] = a1[j]; }
        for (int ct = 0; ct < 4; ++ct) {
            int t = kc * 16 + (cg * 4 + ct);
            v16bf b = *(const v16bf*)(W2p + (size_t)t * 512 + lane * 16);
            acc[ct] = __builtin_amdgcn_wmma_f32_16x16x32_bf16(
                false, a, false, b, (short)0, acc[ct], false, false);
        }
    }
    for (int ct = 0; ct < 4; ++ct) {
        int col = cg * 64 + ct * 16 + m;
        float bias = b2[col];
        for (int r = 0; r < 8; ++r) {
            int row = rs * 16 + hlf * 8 + r;
            out[(size_t)row * 256 + col] = ssilu(acc[ct][r] + bias);
        }
    }
}

extern "C" void kernel_launch(void* const* d_in, const int* in_sizes, int n_in,
                              void* d_out, int out_size, void* d_ws, size_t ws_size,
                              hipStream_t stream) {
    const float* x     = (const float*)d_in[0];
    const int*   batch = (const int*)  d_in[1];
    const float* vx    = (const float*)d_in[2];
    const float* Wn1   = (const float*)d_in[3];
    const float* bn1   = (const float*)d_in[4];
    const float* Wn2   = (const float*)d_in[5];
    const float* bn2   = (const float*)d_in[6];
    const float* Wv1   = (const float*)d_in[7];
    const float* bv1   = (const float*)d_in[8];
    const float* Wv2   = (const float*)d_in[9];
    const float* bv2   = (const float*)d_in[10];

    const int N = in_sizes[0] / 256;   // 200000
    const int B = in_sizes[2] / 256;   // 64

    char* ws = (char*)d_ws;
    __bf16* W1p  = (__bf16*)(ws + 0);        // 16*16*512*2 = 256 KB
    __bf16* W2p  = (__bf16*)(ws + 262144);   //  8*16*512*2 = 128 KB
    __bf16* V1p  = (__bf16*)(ws + 393216);   // 256 KB
    __bf16* V2p  = (__bf16*)(ws + 655360);   // 128 KB
    float*  sums = (float*) (ws + 786432);   // 64*256*4 = 64 KB
    __bf16* Avx  = (__bf16*)(ws + 851968);   // 64*512*2 = 64 KB

    pack_w<<<(16 * 16 * 512 + 255) / 256, 256, 0, stream>>>(Wn1, W1p, 16);
    pack_w<<<( 8 * 16 * 512 + 255) / 256, 256, 0, stream>>>(Wn2, W2p, 8);
    pack_w<<<(16 * 16 * 512 + 255) / 256, 256, 0, stream>>>(Wv1, V1p, 16);
    pack_w<<<( 8 * 16 * 512 + 255) / 256, 256, 0, stream>>>(Wv2, V2p, 8);
    zero_f32<<<(B * 256 + 255) / 256, 256, 0, stream>>>(sums, B * 256);

    node_mlp<<<(N + 31) / 32, 256, 0, stream>>>(x, batch, vx, W1p, bn1, W2p, bn2,
                                                (float*)d_out, sums, N);
    pool_stage<<<1, 256, 0, stream>>>(batch, sums, vx, Avx, N, B);
    vn_mlp<<<1, 512, 0, stream>>>(Avx, V1p, bv1, V2p, bv2,
                                  (float*)d_out + (size_t)N * 256);
}
